// LSTM_14413910245610
// MI455X (gfx1250) — matrix operand order repaired and checked by reading, not compiled, not run on hardware
//
#include <hip/hip_runtime.h>
#include <stdint.h>
#include <stddef.h>

// ---------------- problem constants ----------------
#define T_SEQ   2048
#define BATCH   128
#define HID     512          // STATES_DIM
#define DIM     256          // OUTPUT_DIM
#define NGATE   (4 * HID)    // 2048 gate rows
#define KTOT    (DIM + HID)  // 768 fused K ( [x_in | h] )
#define KSTEPS  (KTOT / 32)  // 24 k-steps for the gate GEMM
#define NWG     32           // persistent workgroups
#define NTHR    256          // threads per workgroup (8 waves)
#define NWAVE   8
#define NWAVES_TOTAL (NWG * NWAVE)   // 256 waves arrive at each barrier

typedef __attribute__((ext_vector_type(16))) __bf16 v16bf;
typedef __attribute__((ext_vector_type(8)))  __bf16 v8bf;
typedef __attribute__((ext_vector_type(8)))  float  v8f;

union FragAB { v16bf f; v8bf h[2]; };

__device__ __forceinline__ uint16_t f2bf(float x) {
    uint32_t u = __float_as_uint(x);
    uint32_t r = u + 0x7FFFu + ((u >> 16) & 1u);   // round-to-nearest-even
    return (uint16_t)(r >> 16);
}

__device__ __forceinline__ float sigmoidf_fast(float x) {
    return 1.0f / (1.0f + __expf(-x));
}

// A-fragment (16x32 bf16): two 8-half runs, 16 halves apart (ISA K-interleave)
__device__ __forceinline__ FragAB load_fragA(const uint16_t* p) {
    FragAB f;
    f.h[0] = *(const v8bf*)(p);
    f.h[1] = *(const v8bf*)(p + 16);
    return f;
}
// B-fragment (32x16 bf16): two 8-half runs, 16 halves apart along K of one weight row
__device__ __forceinline__ FragAB load_fragB(const uint16_t* p) {
    FragAB f;
    f.h[0] = *(const v8bf*)(p);
    f.h[1] = *(const v8bf*)(p + 16);
    return f;
}

// ---------------- prep kernels (parallel, bandwidth bound) ----------------
__global__ void lstm_prep_weights(const float* __restrict__ Wih,
                                  const float* __restrict__ Whh,
                                  const float* __restrict__ bih,
                                  const float* __restrict__ bhh,
                                  const float* __restrict__ wo,
                                  uint16_t* __restrict__ wih_bf,
                                  uint16_t* __restrict__ whh_bf,
                                  uint16_t* __restrict__ wo_bf,
                                  float* __restrict__ bsum,
                                  unsigned* __restrict__ cnt) {
    int i = blockIdx.x * blockDim.x + threadIdx.x;
    int stride = gridDim.x * blockDim.x;
    const int nwih = NGATE * DIM, nwhh = NGATE * HID, nwo = DIM * HID;
    for (int k = i; k < nwhh; k += stride) {
        whh_bf[k] = f2bf(Whh[k]);
        if (k < nwih)  wih_bf[k] = f2bf(Wih[k]);
        if (k < nwo)   wo_bf[k]  = f2bf(wo[k]);
        if (k < NGATE) bsum[k]   = bih[k] + bhh[k];
    }
    if (i == 0) *cnt = 0u;
}

// shifted teacher-forcing input: xbf[0]=0, xbf[t]=bf16(x[t-1])
__global__ void lstm_prep_x(const float* __restrict__ x, uint16_t* __restrict__ xbf) {
    long long i = (long long)blockIdx.x * blockDim.x + threadIdx.x;
    long long stride = (long long)gridDim.x * blockDim.x;
    const long long n = (long long)T_SEQ * BATCH * DIM;
    const long long shift = (long long)BATCH * DIM;
    for (long long k = i; k < n; k += stride)
        xbf[k] = (k < shift) ? (uint16_t)0 : f2bf(x[k - shift]);
}

// -------- wave-level grid barrier (monotonic counter, agent scope) --------
__device__ __forceinline__ void wave_gsync(unsigned* cnt, unsigned target) {
    if ((threadIdx.x & 31) == 0)
        __hip_atomic_fetch_add(cnt, 1u, __ATOMIC_RELEASE, __HIP_MEMORY_SCOPE_AGENT);
    while (__hip_atomic_load(cnt, __ATOMIC_ACQUIRE, __HIP_MEMORY_SCOPE_AGENT) < target)
        __builtin_amdgcn_s_sleep(1);
}

// ---------------- persistent recurrence kernel ----------------
__global__ void __launch_bounds__(NTHR, 1)
lstm_persistent(const uint16_t* __restrict__ xbf,     // (T,B,DIM) bf16, pre-shifted
                const float*    __restrict__ x,       // (T,B,DIM) f32 (for x_pred - x_t)
                const float*    __restrict__ c_init,  // (B,HID)
                const uint16_t* __restrict__ wih_bf,  // (4H,DIM)
                const uint16_t* __restrict__ whh_bf,  // (4H,HID)
                const float*    __restrict__ bsum,    // (4H) = b_ih + b_hh
                const uint16_t* __restrict__ wo_bf,   // (DIM,HID)
                const float*    __restrict__ b_o,     // (DIM)
                uint16_t* __restrict__ hbuf,          // 2 * B*HID bf16 (double buffer)
                unsigned* __restrict__ cnt,
                float* __restrict__ out)              // (T,B,DIM)
{
    extern __shared__ uint16_t smem[];                // 64 rows x KTOT halves = 96 KB
    const int wg   = blockIdx.x;                      // 0..31 : owns channels [wg*16, wg*16+16)
    const int tid  = threadIdx.x;
    const int wave = tid >> 5;                        // 0..7 : owns batch M-tile `wave`
    const int lane = tid & 31;
    const int ln   = lane & 15;
    const int hi   = lane >> 4;

    // ---- stage this WG's 64 gate-weight rows of [W_ih | W_hh] into LDS (bf16)
    for (int e = tid; e < 64 * KTOT; e += NTHR) {
        int r = e / KTOT, k = e - r * KTOT;
        int grow = (r >> 4) * HID + wg * 16 + (r & 15);   // gate g, channel wg*16 + (r&15)
        smem[r * KTOT + k] = (k < DIM) ? wih_bf[grow * DIM + k]
                                       : whh_bf[grow * HID + (k - DIM)];
    }
    __syncthreads();                                  // only intra-WG barrier: LDS ready

    // ---- persistent cell state in VGPRs: c for batch b = wave*16 + r + 8*hi, channel ch
    const int ch = wg * 16 + ln;
    v8f c;
#pragma unroll
    for (int r = 0; r < 8; ++r) {
        int b = wave * 16 + r + 8 * hi;
        c[r] = c_init[b * HID + ch];
    }
    // h0 = tanh(c_init) -> buffer 0
#pragma unroll
    for (int r = 0; r < 8; ++r) {
        int b = wave * 16 + r + 8 * hi;
        hbuf[b * HID + ch] = f2bf(tanhf(c[r]));
    }

    float bias_g[4];
#pragma unroll
    for (int g = 0; g < 4; ++g) bias_g[g] = bsum[g * HID + ch];

    // LDS row base pointers for the 4 gate B-tiles of this lane
    const uint16_t* srow0 = smem + (0 * 16 + ln) * KTOT + hi * 8;
    const uint16_t* srow1 = smem + (1 * 16 + ln) * KTOT + hi * 8;
    const uint16_t* srow2 = smem + (2 * 16 + ln) * KTOT + hi * 8;
    const uint16_t* srow3 = smem + (3 * 16 + ln) * KTOT + hi * 8;

    // phase-B (output GEMM) ownership: first 128 global waves, one 16x16 tile each
    const int  gwid = wg * NWAVE + wave;
    const bool doB  = (gwid < 128);
    const int  mtB  = gwid >> 4, ntB = gwid & 15;
    const float boB = doB ? b_o[ntB * 16 + ln] : 0.0f;
    const uint16_t* worow = wo_bf + (ntB * 16 + ln) * HID + hi * 8;

    unsigned bno = 1;
    wave_gsync(cnt, bno * NWAVES_TOTAL);              // h0 visible everywhere

    for (int t = 0; t < T_SEQ; ++t) {
        const uint16_t* hcur = hbuf + (size_t)(t & 1) * (BATCH * HID);
        uint16_t*       hnxt = hbuf + (size_t)((t & 1) ^ 1) * (BATCH * HID);

        // ---- HBM latency hiding: prefetch next step's x_in rows (bf16) and this
        // step's fp32 x rows for the phase-B tile while the gate GEMM runs.
        if (t + 1 < T_SEQ) {
            const char* nx = (const char*)(xbf + (size_t)(t + 1) * BATCH * DIM
                                               + (size_t)(wave * 16 + ln) * DIM);
            __builtin_prefetch(nx + hi * 256,       0, 3);
            __builtin_prefetch(nx + hi * 256 + 128, 0, 3);
        }
        if (doB) {
            const char* px = (const char*)(x + (size_t)t * BATCH * DIM
                                             + (size_t)(mtB * 16 + ln) * DIM);
            __builtin_prefetch(px + hi * 512,       0, 3);
            __builtin_prefetch(px + hi * 512 + 128, 0, 3);
            __builtin_prefetch(px + hi * 512 + 256, 0, 3);
            __builtin_prefetch(px + hi * 512 + 384, 0, 3);
        }

        // ===== Phase A: gates = [x_in | h] @ [W_ih | W_hh]^T + bias =====
        v8f acc0, acc1, acc2, acc3;
#pragma unroll
        for (int r = 0; r < 8; ++r) {
            acc0[r] = bias_g[0]; acc1[r] = bias_g[1];
            acc2[r] = bias_g[2]; acc3[r] = bias_g[3];
        }
        const int am = wave * 16 + ln;                       // A-matrix row (batch index)
        const uint16_t* aX = xbf + (size_t)t * BATCH * DIM + am * DIM + hi * 8;  // k in [0,DIM)
        const uint16_t* aH = hcur + am * HID + hi * 8;                           // index k-DIM

        // software-pipelined, double-buffered fragments (renamed away by unroll)
        FragAB Ac  = load_fragA(aX);
        FragAB Bc0 = load_fragB(srow0);
        FragAB Bc1 = load_fragB(srow1);
        FragAB Bc2 = load_fragB(srow2);
        FragAB Bc3 = load_fragB(srow3);

#pragma unroll
        for (int ks = 0; ks < KSTEPS; ++ks) {
            FragAB An, Bn0, Bn1, Bn2, Bn3;
            if (ks + 1 < KSTEPS) {                           // issue next k-step's loads first
                const int kn = (ks + 1) * 32;
                const uint16_t* ap = (kn < DIM) ? (aX + kn) : (aH + (kn - DIM));
                An  = load_fragA(ap);
                Bn0 = load_fragB(srow0 + kn);
                Bn1 = load_fragB(srow1 + kn);
                Bn2 = load_fragB(srow2 + kn);
                Bn3 = load_fragB(srow3 + kn);
            }
            acc0 = __builtin_amdgcn_wmma_f32_16x16x32_bf16(
                false, Ac.f, false, Bc0.f, (short)0, acc0, false, false);
            acc1 = __builtin_amdgcn_wmma_f32_16x16x32_bf16(
                false, Ac.f, false, Bc1.f, (short)0, acc1, false, false);
            acc2 = __builtin_amdgcn_wmma_f32_16x16x32_bf16(
                false, Ac.f, false, Bc2.f, (short)0, acc2, false, false);
            acc3 = __builtin_amdgcn_wmma_f32_16x16x32_bf16(
                false, Ac.f, false, Bc3.f, (short)0, acc3, false, false);
            if (ks + 1 < KSTEPS) {
                Ac = An; Bc0 = Bn0; Bc1 = Bn1; Bc2 = Bn2; Bc3 = Bn3;
            }
        }

        // ===== cell update (lanes line up: same (b, ch) across the 4 gate accs) =====
#pragma unroll
        for (int r = 0; r < 8; ++r) {
            float ig = sigmoidf_fast(acc0[r]);
            float fg = sigmoidf_fast(acc1[r]);
            float gg = tanhf(acc2[r]);
            float og = sigmoidf_fast(acc3[r]);
            float cn = fg * c[r] + ig * gg;
            c[r] = cn;
            float hn = og * tanhf(cn);
            int b = wave * 16 + r + 8 * hi;
            hnxt[b * HID + ch] = f2bf(hn);
        }

        // ===== grid barrier (wave-level): h(t+1) complete & visible =====
        ++bno;
        wave_gsync(cnt, bno * NWAVES_TOTAL);

        // ===== Phase B: out[t] = h_new @ w_o^T + b_o - x[t] =====
        if (doB) {
            v8f o;
#pragma unroll
            for (int r = 0; r < 8; ++r) o[r] = boB;
            const uint16_t* ah = hnxt + (mtB * 16 + ln) * HID + hi * 8;

            FragAB Ac2 = load_fragA(ah);
            FragAB Bc2_ = load_fragB(worow);
#pragma unroll
            for (int ks = 0; ks < HID / 32; ++ks) {
                FragAB An2, Bn2_;
                if (ks + 1 < HID / 32) {
                    const int kn = (ks + 1) * 32;
                    An2  = load_fragA(ah + kn);
                    Bn2_ = load_fragB(worow + kn);
                }
                o = __builtin_amdgcn_wmma_f32_16x16x32_bf16(
                    false, Ac2.f, false, Bc2_.f, (short)0, o, false, false);
                if (ks + 1 < HID / 32) { Ac2 = An2; Bc2_ = Bn2_; }
            }
            const float* xt = x   + (size_t)t * BATCH * DIM;
            float*       ot = out + (size_t)t * BATCH * DIM;
#pragma unroll
            for (int r = 0; r < 8; ++r) {
                int b = mtB * 16 + r + 8 * hi;
                int n = ntB * 16 + ln;
                ot[b * DIM + n] = o[r] - xt[b * DIM + n];
            }
        }
    }
}

// ---------------- host entry ----------------
extern "C" void kernel_launch(void* const* d_in, const int* in_sizes, int n_in,
                              void* d_out, int out_size, void* d_ws, size_t ws_size,
                              hipStream_t stream) {
    (void)in_sizes; (void)n_in; (void)out_size; (void)ws_size;

    const float* x      = (const float*)d_in[0];   // (T,B,DIM)
    const float* c_init = (const float*)d_in[1];   // (B,HID)
    const float* W_ih   = (const float*)d_in[2];   // (4H,DIM)
    const float* W_hh   = (const float*)d_in[3];   // (4H,HID)
    const float* b_ih   = (const float*)d_in[4];   // (4H)
    const float* b_hh   = (const float*)d_in[5];   // (4H)
    const float* w_o    = (const float*)d_in[6];   // (DIM,HID)
    const float* b_o    = (const float*)d_in[7];   // (DIM)
    float* out = (float*)d_out;

    // workspace carve-up (all offsets 256B aligned)
    char* ws = (char*)d_ws;
    size_t off = 0;
    unsigned* cnt    = (unsigned*)(ws + off); off += 256;
    float*    bsum   = (float*)   (ws + off); off += (size_t)NGATE * 4;            // 8 KB
    uint16_t* hbuf   = (uint16_t*)(ws + off); off += (size_t)2 * BATCH * HID * 2;  // 256 KB
    uint16_t* wih_bf = (uint16_t*)(ws + off); off += (size_t)NGATE * DIM * 2;      // 1 MB
    uint16_t* whh_bf = (uint16_t*)(ws + off); off += (size_t)NGATE * HID * 2;      // 2 MB
    uint16_t* wo_bf  = (uint16_t*)(ws + off); off += (size_t)DIM * HID * 2;        // 256 KB
    uint16_t* xbf    = (uint16_t*)(ws + off); off += (size_t)T_SEQ * BATCH * DIM * 2; // 128 MB

    lstm_prep_weights<<<dim3(2048), dim3(256), 0, stream>>>(
        W_ih, W_hh, b_ih, b_hh, w_o, wih_bf, whh_bf, wo_bf, bsum, cnt);

    lstm_prep_x<<<dim3(4096), dim3(256), 0, stream>>>(x, xbf);

    static const size_t LDS_BYTES = (size_t)64 * KTOT * 2;   // 96 KB
    hipFuncSetAttribute((const void*)lstm_persistent,
                        hipFuncAttributeMaxDynamicSharedMemorySize, (int)LDS_BYTES);

    lstm_persistent<<<dim3(NWG), dim3(NTHR), LDS_BYTES, stream>>>(
        xbf, x, c_init, wih_bf, whh_bf, bsum, wo_bf, b_o, hbuf, cnt, out);
}
